// BClassifier_91139206021415
// MI455X (gfx1250) — matrix operand order repaired and checked by reading, not compiled, not run on hardware
//
#include <hip/hip_runtime.h>
#include <stdint.h>

// ---------------------------------------------------------------------------
// SNN classifier for MI455X (gfx1250, wave32, WMMA).
//   161 GFLOP GEMM (M=3200,N=2048,K=12288) -> bf16 WMMA, f32 accumulate,
//   double-buffered LDS fed by async global->LDS loads (ASYNCcnt).
//   Then a serial T=25 LIF scan (1 block / batch element).
// ---------------------------------------------------------------------------

typedef __bf16 bf16_t;
typedef bf16_t v16bf __attribute__((ext_vector_type(16)));
typedef bf16_t v8bf  __attribute__((ext_vector_type(8)));
typedef bf16_t v4bf  __attribute__((ext_vector_type(4)));
typedef float  v8f   __attribute__((ext_vector_type(8)));

// gcc-style int4 vector, as required by the async-LDS builtin prototype
typedef int v4i __attribute__((vector_size(16)));
typedef v4i __attribute__((address_space(1))) gv4i;   // global int4
typedef v4i __attribute__((address_space(3))) sv4i;   // LDS int4

// Problem sizes (fixed by the reference)
constexpr int BATCH = 128;
constexpr int TSTEP = 25;
constexpr int KDIM  = 12288;           // F = 3*64*64
constexpr int NDIM  = 2048;            // HID
constexpr int MDIM  = BATCH * TSTEP;   // 3200

// GEMM tiling
constexpr int BM  = 128;
constexpr int BN  = 128;
constexpr int BK  = 64;                // two WMMA k-steps per LDS tile
constexpr int LDK = BK + 8;            // padded row stride: 144B = 36 banks
                                       //  -> conflict-free b128 fragment loads
constexpr int NITER = KDIM / BK;       // 192

#if __has_builtin(__builtin_amdgcn_global_load_async_to_lds_b128) && \
    __has_builtin(__builtin_amdgcn_s_wait_asynccnt)
#define USE_ASYNC 1
#else
#define USE_ASYNC 0
#endif

// ---------------------------------------------------------------------------
// f32 -> bf16 conversion (vectorized, grid-stride)
// ---------------------------------------------------------------------------
__global__ void cvt_f32_bf16(const float* __restrict__ src,
                             bf16_t* __restrict__ dst, int n4) {
    int stride = gridDim.x * blockDim.x;
    for (int i = blockIdx.x * blockDim.x + threadIdx.x; i < n4; i += stride) {
        float4 v = reinterpret_cast<const float4*>(src)[i];
        v4bf o;
        o[0] = (bf16_t)v.x; o[1] = (bf16_t)v.y;
        o[2] = (bf16_t)v.z; o[3] = (bf16_t)v.w;
        reinterpret_cast<v4bf*>(dst)[i] = o;
    }
}

// ---------------------------------------------------------------------------
// Tile staging: 2 threads per tile row, 32 bf16 (64B, four b128 chunks) each.
// ---------------------------------------------------------------------------
#if USE_ASYNC
template <int OFF>
__device__ __forceinline__ void async_b128(const bf16_t* g, bf16_t* s) {
    // INST_OFFSET is added to BOTH the global and the LDS address (ISA 10.7),
    // so one address pair serves all four 16B chunks.
    __builtin_amdgcn_global_load_async_to_lds_b128(
        (gv4i*)const_cast<bf16_t*>(g), (sv4i*)s, OFF, 0);
}
#endif

__device__ __forceinline__ void load_tile(const bf16_t* ga, const bf16_t* gb,
                                          bf16_t* sa, bf16_t* sb) {
#if USE_ASYNC
    async_b128<0>(ga, sa);  async_b128<16>(ga, sa);
    async_b128<32>(ga, sa); async_b128<48>(ga, sa);
    async_b128<0>(gb, sb);  async_b128<16>(gb, sb);
    async_b128<32>(gb, sb); async_b128<48>(gb, sb);
#else
    const uint4* ga4 = reinterpret_cast<const uint4*>(ga);
    const uint4* gb4 = reinterpret_cast<const uint4*>(gb);
    uint4 a0 = ga4[0], a1 = ga4[1], a2 = ga4[2], a3 = ga4[3];
    uint4 b0 = gb4[0], b1 = gb4[1], b2 = gb4[2], b3 = gb4[3];
    uint4* s4a = reinterpret_cast<uint4*>(sa);
    uint4* s4b = reinterpret_cast<uint4*>(sb);
    s4a[0] = a0; s4a[1] = a1; s4a[2] = a2; s4a[3] = a3;
    s4b[0] = b0; s4b[1] = b1; s4b[2] = b2; s4b[3] = b3;
#endif
}

// ---------------------------------------------------------------------------
// h_all[m, n] = sum_k A[m,k] * W1[n,k] + b1[n]
//   A: [MDIM, KDIM] bf16 (row m = b*T + t), W1: [NDIM, KDIM] bf16.
// 256 threads = 8 waves; wave grid 2(m) x 4(n); each wave 4x2 16x16 tiles.
// Double-buffered LDS; async global->LDS copy overlapped with WMMA.
// ---------------------------------------------------------------------------
__global__ __launch_bounds__(256)
void gemm_h(const bf16_t* __restrict__ A,
            const bf16_t* __restrict__ W1b,
            const float*  __restrict__ b1,
            float* __restrict__ Hout) {
    __shared__ __align__(16) bf16_t a_lds[2][BM * LDK];
    __shared__ __align__(16) bf16_t b_lds[2][BN * LDK];

    const int tid     = threadIdx.x;
    const int lane    = tid & 31;
    const int w       = tid >> 5;
    const int wave_m  = w >> 2;        // 0..1  -> 64 rows each
    const int wave_n  = w & 3;         // 0..3  -> 32 cols each
    const int lane_lo = lane & 15;
    const int hi      = lane >> 4;     // 0 or 1

    const int blockN = blockIdx.x * BN;
    const int blockM = blockIdx.y * BM;

    const int lr = tid >> 1;           // tile row 0..127
    const int lh = tid & 1;            // which 32-element half of BK
    const bf16_t* gA = A   + (size_t)(blockM + lr) * KDIM + lh * 32;
    const bf16_t* gB = W1b + (size_t)(blockN + lr) * KDIM + lh * 32;
    const int soff = lr * LDK + lh * 32;

    v8f acc[4][2];
    const v8f zero = {0.f, 0.f, 0.f, 0.f, 0.f, 0.f, 0.f, 0.f};
#pragma unroll
    for (int mt = 0; mt < 4; ++mt)
#pragma unroll
        for (int nt = 0; nt < 2; ++nt) acc[mt][nt] = zero;

    // Prime buffer 0
    load_tile(gA, gB, &a_lds[0][soff], &b_lds[0][soff]);

    for (int it = 0; it < NITER; ++it) {
        const int cur = it & 1;

        if (it + 1 < NITER) {
            const int kk = (it + 1) * BK;
            load_tile(gA + kk, gB + kk,
                      &a_lds[cur ^ 1][soff], &b_lds[cur ^ 1][soff]);
            if (kk + BK < KDIM) {      // keep L2 warm one more tile ahead
                __builtin_prefetch(gA + kk + BK, 0, 1);
                __builtin_prefetch(gB + kk + BK, 0, 1);
            }
#if USE_ASYNC
            // 8 newer ops in flight; in-order retirement => tile `cur` landed
            __builtin_amdgcn_s_wait_asynccnt(8);
#endif
        } else {
#if USE_ASYNC
            __builtin_amdgcn_s_wait_asynccnt(0);
#endif
        }
        __syncthreads();

        const bf16_t* al = a_lds[cur];
        const bf16_t* bl = b_lds[cur];
#pragma unroll
        for (int ks = 0; ks < BK; ks += 32) {
            // A fragments (16x32 bf16, ISA 7.12.2): lane_lo = M row; lane-hi
            // half holds k-chunks {hi*8..+7} and {16+hi*8..+7}.
            v16bf afrag[4];
#pragma unroll
            for (int mt = 0; mt < 4; ++mt) {
                int row = wave_m * 64 + mt * 16 + lane_lo;
                v8bf lo = *reinterpret_cast<const v8bf*>(
                    &al[row * LDK + ks + hi * 8]);
                v8bf hi8 = *reinterpret_cast<const v8bf*>(
                    &al[row * LDK + ks + 16 + hi * 8]);
#pragma unroll
                for (int e = 0; e < 8; ++e) {
                    afrag[mt][e] = lo[e];
                    afrag[mt][e + 8] = hi8[e];
                }
            }
            // B fragments (32x16 bf16): lane_lo = N col, lane-hi half holds
            // k-chunks {hi*8..+7} and {16+hi*8..+7} (same as A).
            v16bf bfrag[2];
#pragma unroll
            for (int nt = 0; nt < 2; ++nt) {
                int col = wave_n * 32 + nt * 16 + lane_lo;
                const v8bf* p = reinterpret_cast<const v8bf*>(
                    &bl[col * LDK + ks + hi * 8]);
                v8bf lo = p[0], hi8 = p[2];
#pragma unroll
                for (int e = 0; e < 8; ++e) {
                    bfrag[nt][e] = lo[e];
                    bfrag[nt][e + 8] = hi8[e];
                }
            }
#pragma unroll
            for (int mt = 0; mt < 4; ++mt)
#pragma unroll
                for (int nt = 0; nt < 2; ++nt)
                    acc[mt][nt] = __builtin_amdgcn_wmma_f32_16x16x32_bf16(
                        false, afrag[mt], false, bfrag[nt],
                        (short)0, acc[mt][nt], false, false);
        }
        __syncthreads();
    }

    // Epilogue: C layout — VGPR g holds row (hi*8 + g), col = lane_lo.
#pragma unroll
    for (int nt = 0; nt < 2; ++nt) {
        int col = blockN + wave_n * 32 + nt * 16 + lane_lo;
        float bias = b1[col];
#pragma unroll
        for (int mt = 0; mt < 4; ++mt) {
            int row0 = blockM + wave_m * 64 + mt * 16 + hi * 8;
#pragma unroll
            for (int g = 0; g < 8; ++g)
                Hout[(size_t)(row0 + g) * NDIM + col] = acc[mt][nt][g] + bias;
        }
    }
}

// ---------------------------------------------------------------------------
// LIF scan: one block per batch element. 256 threads x 8 hidden units.
// ---------------------------------------------------------------------------
__global__ __launch_bounds__(256)
void snn_scan(const float* __restrict__ Hall,   // [MDIM, NDIM], m = b*T + t
              const float* __restrict__ Wo,     // [2, NDIM]
              const float* __restrict__ bo,     // [2]
              float* __restrict__ out) {        // [BATCH, 2]
    const int b   = blockIdx.x;
    const int tid = threadIdx.x;
    const int hbase = tid * 8;

    float wo0[8], wo1[8], mem1[8];
#pragma unroll
    for (int j = 0; j < 8; ++j) {
        wo0[j]  = Wo[hbase + j];
        wo1[j]  = Wo[NDIM + hbase + j];
        mem1[j] = 0.f;
    }
    const float bo0 = bo[0], bo1 = bo[1];

    __shared__ float red0[8];
    __shared__ float red1[8];
    float memo0 = 0.f, memo1 = 0.f, sum0 = 0.f, sum1 = 0.f;

    for (int t = 0; t < TSTEP; ++t) {
        const float* hrow = Hall + (size_t)(b * TSTEP + t) * NDIM;
        float p0 = 0.f, p1 = 0.f;
#pragma unroll
        for (int j = 0; j < 8; ++j) {
            float h = hrow[hbase + j];
            float m = mem1[j];
            float r1 = (m > 1.0f) ? 1.0f : 0.0f;   // reset from previous mem
            m = 0.9f * m + h - r1;                 // THR == 1
            mem1[j] = m;
            if (m > 1.0f) { p0 += wo0[j]; p1 += wo1[j]; }
        }
#pragma unroll
        for (int off = 16; off > 0; off >>= 1) {
            p0 += __shfl_xor(p0, off, 32);
            p1 += __shfl_xor(p1, off, 32);
        }
        if ((tid & 31) == 0) { red0[tid >> 5] = p0; red1[tid >> 5] = p1; }
        __syncthreads();
        if (tid == 0) {
            float o0 = bo0, o1 = bo1;
#pragma unroll
            for (int ww = 0; ww < 8; ++ww) { o0 += red0[ww]; o1 += red1[ww]; }
            float r0 = (memo0 > 1.0f) ? 1.0f : 0.0f;
            float r1 = (memo1 > 1.0f) ? 1.0f : 0.0f;
            memo0 = 0.9f * memo0 + o0 - r0;
            memo1 = 0.9f * memo1 + o1 - r1;
            if (memo0 > 1.0f) sum0 += 1.0f;
            if (memo1 > 1.0f) sum1 += 1.0f;
        }
        __syncthreads();
    }
    if (tid == 0) {
        out[b * 2 + 0] = sum0;
        out[b * 2 + 1] = sum1;
    }
}

// ---------------------------------------------------------------------------
extern "C" void kernel_launch(void* const* d_in, const int* in_sizes, int n_in,
                              void* d_out, int out_size, void* d_ws, size_t ws_size,
                              hipStream_t stream) {
    const float* x  = (const float*)d_in[0];   // [B, T, C, H, W] f32
    const float* W1 = (const float*)d_in[1];   // [HID, F]
    const float* b1 = (const float*)d_in[2];   // [HID]
    const float* Wo = (const float*)d_in[3];   // [2, HID]
    const float* bo = (const float*)d_in[4];   // [2]
    float* out = (float*)d_out;                // [B, 2]

    // Workspace layout (all offsets 256B-aligned):
    //   xb   : MDIM*KDIM bf16  (78.6 MB)
    //   w1b  : NDIM*KDIM bf16  (50.3 MB)  -- fits in the 192MB L2
    //   hall : MDIM*NDIM f32   (26.2 MB)
    char* ws = (char*)d_ws;
    bf16_t* xb   = (bf16_t*)ws;
    bf16_t* w1b  = (bf16_t*)(ws + (size_t)MDIM * KDIM * 2);
    float*  hall = (float*)(ws + (size_t)MDIM * KDIM * 2
                               + (size_t)NDIM * KDIM * 2);

    cvt_f32_bf16<<<2048, 256, 0, stream>>>(x,  xb,  MDIM * KDIM / 4);
    cvt_f32_bf16<<<2048, 256, 0, stream>>>(W1, w1b, NDIM * KDIM / 4);

    dim3 grid(NDIM / BN, MDIM / BM);   // 16 x 25
    gemm_h<<<grid, 256, 0, stream>>>(xb, w1b, b1, hall);

    snn_scan<<<BATCH, 256, 0, stream>>>(hall, Wo, bo, out);

    (void)in_sizes; (void)n_in; (void)out_size; (void)ws_size;
}
